// Hgru1_11055245820546
// MI455X (gfx1250) — matrix operand order repaired and checked by reading, not compiled, not run on hardware
//
#include <hip/hip_runtime.h>
#include <cstdint>

// ---------------- problem constants (match reference) ----------------
#define BB   4
#define NNv  2048
#define EEv  2048
#define HDD  128
#define EE2  4096
#define BNv  (BB * NNv)          // 8192 rows
#define EPSV 1e-6f

typedef __bf16 bf16_t;
typedef __attribute__((ext_vector_type(16))) __bf16 bf16x16;
typedef __attribute__((ext_vector_type(8)))  __bf16 bf16x8;
typedef __attribute__((ext_vector_type(8)))  float  f32x8;

// LDS tile geometry: K-slab of 32 bf16 per row, padded to 40 (80B; 16B aligned,
// and 16 row-strided lanes hit distinct banks: stride 20 dwords, gcd(20,64)=4 -> period 16)
#define LROW 40

// round-to-nearest-even f32 -> bf16
__device__ __forceinline__ bf16_t f2bf(float f) {
  unsigned u = __builtin_bit_cast(unsigned, f);
  u += 0x7FFFu + ((u >> 16) & 1u);
  unsigned short h = (unsigned short)(u >> 16);
  return __builtin_bit_cast(bf16_t, h);
}

// ---------------- CDNA5 async global->LDS copy (ASYNCcnt path) ----------------
// 16 bytes per lane: LDS dest offset in a VGPR, 64-bit global address in a VGPR pair.
__device__ __forceinline__ void async_cp16(const bf16_t* g, const bf16_t* l) {
  uint32_t lo = (uint32_t)(uintptr_t)l;              // generic LDS addr low 32 = LDS offset
  unsigned long long ga = (unsigned long long)(uintptr_t)g;
  asm volatile("global_load_async_to_lds_b128 %0, %1, off"
               :: "v"(lo), "v"(ga) : "memory");
}
__device__ __forceinline__ void wait_async0() {
  asm volatile("s_wait_asynccnt 0" ::: "memory");
}

// ---------------- elementwise cast / transpose-cast ----------------
__global__ void k_cast_bf16(const float* __restrict__ s, bf16_t* __restrict__ d, int n) {
  int i = blockIdx.x * blockDim.x + threadIdx.x;
  if (i < n) d[i] = f2bf(s[i]);
}

// src: [R][C] f32 row-major  ->  dst: [C][R] bf16 row-major
__global__ void k_transpose_bf16(const float* __restrict__ s, bf16_t* __restrict__ d,
                                 int R, int C) {
  int i = blockIdx.x * blockDim.x + threadIdx.x;
  if (i < R * C) {
    int r = i / C, c = i - r * C;
    d[(size_t)c * R + r] = f2bf(s[i]);
  }
}

// ---------------- WMMA GEMM: C[M x N] = act( A[M x K] * BT[N x K]^T ) ----------------
// Block = 256 threads = 8 waves arranged 2(M) x 4(N); block tile 64 x 128.
// Per K=32 slab: A(64x32) + B(128x32) staged to LDS with async copies (double-buffered),
// waves read fragments via ds_load_b128 and issue 4x v_wmma each.
enum { ACT_NONE = 0, ACT_SILU = 1, ACT_GATE = 2 };

template <int ACT, bool OUT_BF>
__device__ __forceinline__ void store_tile(const f32x8& c, void* Cout,
                                           int mBase, int nCol, int N) {
#pragma unroll
  for (int r = 0; r < 8; ++r) {
    float y = c[r];
    if (ACT == ACT_SILU) {
      y = y / (1.f + __expf(-y));                       // silu
    } else if (ACT == ACT_GATE) {
      float f = 1.f / (1.f + __expf(-y));               // f = sigmoid(raw)
      y = 1.f / (1.f + __expf(-f));                     // a = sigmoid(f)
    }
    size_t idx = (size_t)(mBase + r) * (size_t)N + (size_t)nCol;
    if (OUT_BF) ((bf16_t*)Cout)[idx] = f2bf(y);
    else        ((float*)Cout)[idx]  = y;
  }
}

// lo 8 bf16 at p+o1, hi 8 bf16 at p+o2 -> one 16-element WMMA operand
__device__ __forceinline__ bf16x16 ld_frag(const bf16_t* p, int o1, int o2) {
  bf16x8 lo = *(const bf16x8*)(p + o1);
  bf16x8 hi = *(const bf16x8*)(p + o2);
  bf16x16 r;
#pragma unroll
  for (int i = 0; i < 8; ++i) { r[i] = lo[i]; r[i + 8] = hi[i]; }
  return r;
}

// Stage ROWS x 32 bf16 slab (global, row stride ldg) into LDS (row stride LROW).
template <int ROWS>
__device__ __forceinline__ void stage_tile(const bf16_t* __restrict__ G, int ldg,
                                           bf16_t* lbase, int kcur, int tid) {
#pragma unroll
  for (int rr = 0; rr < ROWS / 64; ++rr) {
    int row   = (tid >> 2) + rr * 64;    // 256 threads -> 64 rows x 4 chunks
    int chunk = tid & 3;                 // 4 x 8 bf16 = 32 k
    async_cp16(G + (size_t)row * ldg + kcur + chunk * 8,
               lbase + row * LROW + chunk * 8);
  }
}

template <int ACT, bool OUT_BF>
__global__ void __launch_bounds__(256)
k_wmma_gemm(const bf16_t* __restrict__ A, const bf16_t* __restrict__ BT,
            void* __restrict__ Cout, int M, int N, int K) {
  __shared__ alignas(16) bf16_t aTile[2][64 * LROW];    // 10 KB
  __shared__ alignas(16) bf16_t bTile[2][128 * LROW];   // 20 KB

  const int tid  = threadIdx.x;
  const int lane = tid & 31;
  const int l16  = lane & 15;
  const int hb   = lane >> 4;            // half-wave K sub-striping
  const int wv   = tid >> 5;             // 0..7
  const int wM   = (wv >> 2) * 32;       // wave M offset in block tile
  const int wN   = (wv & 3) * 32;        // wave N offset in block tile

  const int tilesNb = N >> 7;            // block tile N = 128
  const int tMB = blockIdx.x / tilesNb;
  const int tNB = blockIdx.x - tMB * tilesNb;
  const int m0B = tMB << 6, n0B = tNB << 7;
  if (m0B >= M) return;                  // block-uniform

  const bf16_t* Ag = A  + (size_t)m0B * K;
  const bf16_t* Bg = BT + (size_t)n0B * K;

  f32x8 c00 = {}, c01 = {}, c10 = {}, c11 = {};

  const int KT = K >> 5;
  stage_tile<64 >(Ag, K, &aTile[0][0], 0, tid);
  stage_tile<128>(Bg, K, &bTile[0][0], 0, tid);

  for (int kt = 0; kt < KT; ++kt) {
    wait_async0();                       // my stage-kt copies landed in LDS
    __syncthreads();                     // everyone's landed; prev compute done
    if (kt + 1 < KT) {                   // overlap next slab with this compute
      stage_tile<64 >(Ag, K, &aTile[(kt + 1) & 1][0], (kt + 1) << 5, tid);
      stage_tile<128>(Bg, K, &bTile[(kt + 1) & 1][0], (kt + 1) << 5, tid);
    }
    const bf16_t* aL = &aTile[kt & 1][0];
    const bf16_t* bL = &bTile[kt & 1][0];

    bf16x16 a0 = ld_frag(aL + (wM + l16)      * LROW, hb * 8,  hb * 8 + 16);
    bf16x16 a1 = ld_frag(aL + (wM + 16 + l16) * LROW, hb * 8,  hb * 8 + 16);
    bf16x16 b0 = ld_frag(bL + (wN + l16)      * LROW, hb * 8,  hb * 8 + 16);
    bf16x16 b1 = ld_frag(bL + (wN + 16 + l16) * LROW, hb * 8,  hb * 8 + 16);

    c00 = __builtin_amdgcn_wmma_f32_16x16x32_bf16(false, a0, false, b0, (short)0, c00, false, false);
    c01 = __builtin_amdgcn_wmma_f32_16x16x32_bf16(false, a0, false, b1, (short)0, c01, false, false);
    c10 = __builtin_amdgcn_wmma_f32_16x16x32_bf16(false, a1, false, b0, (short)0, c10, false, false);
    c11 = __builtin_amdgcn_wmma_f32_16x16x32_bf16(false, a1, false, b1, (short)0, c11, false, false);
  }

  const int m0 = m0B + wM, n0 = n0B + wN;
  store_tile<ACT, OUT_BF>(c00, Cout, m0 + hb * 8,      n0 + l16,      N);
  store_tile<ACT, OUT_BF>(c01, Cout, m0 + hb * 8,      n0 + 16 + l16, N);
  store_tile<ACT, OUT_BF>(c10, Cout, m0 + 16 + hb * 8, n0 + l16,      N);
  store_tile<ACT, OUT_BF>(c11, Cout, m0 + 16 + hb * 8, n0 + 16 + l16, N);
}

// ---------------- HGRN-1 gated scan: h_t = a_t*h_{t-1} + (1-a_t)*v_t; Qb <- q*h ----------------
__global__ void k_hgru_scan(float* __restrict__ Qb, const float* __restrict__ Ab,
                            const float* __restrict__ Vb) {
  int c = blockIdx.x * blockDim.x + threadIdx.x;    // one thread per (batch, channel)
  if (c >= BB * EE2) return;
  int b  = c / EE2;
  int ch = c - b * EE2;
  size_t base = (size_t)b * NNv * EE2 + ch;
  float h = 0.f;
  for (int t = 0; t < NNv; ++t) {
    size_t i = base + (size_t)t * EE2;              // coalesced across threads
    float a = Ab[i];
    float v = Vb[i];
    float q = Qb[i];
    h = fmaf(a, h, (1.f - a) * v);
    Qb[i] = q * h;                                  // out = q * h (in place)
  }
}

// ---------------- RMSNorm over channel dim (4096), output bf16 ----------------
__global__ void k_rmsnorm_bf16(const float* __restrict__ X, const float* __restrict__ w,
                               bf16_t* __restrict__ Y) {
  __shared__ float red[8];
  __shared__ float scale_s;
  size_t row = blockIdx.x;
  const float* x = X + row * EE2;
  float s = 0.f;
  for (int i = threadIdx.x; i < EE2; i += blockDim.x) { float v = x[i]; s += v * v; }
#pragma unroll
  for (int off = 16; off > 0; off >>= 1) s += __shfl_xor(s, off, 32);   // wave32 reduce
  if ((threadIdx.x & 31) == 0) red[threadIdx.x >> 5] = s;
  __syncthreads();
  if (threadIdx.x == 0) {
    float tot = 0.f;
#pragma unroll
    for (int i = 0; i < 8; ++i) tot += red[i];
    scale_s = rsqrtf(tot / (float)EE2 + EPSV);
  }
  __syncthreads();
  float scale = scale_s;
  bf16_t* y = Y + row * EE2;
  for (int i = threadIdx.x; i < EE2; i += blockDim.x) y[i] = f2bf(x[i] * scale * w[i]);
}

// ---------------- launch ----------------
extern "C" void kernel_launch(void* const* d_in, const int* in_sizes, int n_in,
                              void* d_out, int out_size, void* d_ws, size_t ws_size,
                              hipStream_t stream) {
  (void)in_sizes; (void)n_in; (void)out_size; (void)ws_size;
  const float* x   = (const float*)d_in[0];
  const float* Wq1 = (const float*)d_in[1];
  const float* Wq2 = (const float*)d_in[2];
  const float* Wk1 = (const float*)d_in[3];
  const float* Wk2 = (const float*)d_in[4];
  const float* Wv  = (const float*)d_in[5];
  const float* Wo  = (const float*)d_in[6];
  const float* nw  = (const float*)d_in[7];

  char* p = (char*)d_ws;
  auto take = [&](size_t bytes) -> char* {
    char* r = p; p += (bytes + 255) & ~(size_t)255; return r;
  };
  bf16_t* xbf  = (bf16_t*)take((size_t)BNv * EEv * 2);
  bf16_t* wq1t = (bf16_t*)take((size_t)HDD * EEv * 2);
  bf16_t* wk1t = (bf16_t*)take((size_t)HDD * EEv * 2);
  bf16_t* wq2t = (bf16_t*)take((size_t)EE2 * HDD * 2);
  bf16_t* wk2t = (bf16_t*)take((size_t)EE2 * HDD * 2);
  bf16_t* wvt  = (bf16_t*)take((size_t)EE2 * EEv * 2);
  bf16_t* wot  = (bf16_t*)take((size_t)EEv * EE2 * 2);
  bf16_t* tq   = (bf16_t*)take((size_t)BNv * HDD * 2);
  bf16_t* tk   = (bf16_t*)take((size_t)BNv * HDD * 2);
  float*  Qb   = (float*)take((size_t)BNv * EE2 * 4);
  float*  Ab   = (float*)take((size_t)BNv * EE2 * 4);
  float*  Vb   = (float*)take((size_t)BNv * EE2 * 4);
  bf16_t* obf  = (bf16_t*)take((size_t)BNv * EE2 * 2);

  dim3 blk(256);
  // casts + weight transposes (weights become BT = [N][K] bf16)
  k_cast_bf16<<<(BNv * EEv + 255) / 256, blk, 0, stream>>>(x, xbf, BNv * EEv);
  k_transpose_bf16<<<(EEv * HDD + 255) / 256, blk, 0, stream>>>(Wq1, wq1t, EEv, HDD);
  k_transpose_bf16<<<(EEv * HDD + 255) / 256, blk, 0, stream>>>(Wk1, wk1t, EEv, HDD);
  k_transpose_bf16<<<(HDD * EE2 + 255) / 256, blk, 0, stream>>>(Wq2, wq2t, HDD, EE2);
  k_transpose_bf16<<<(HDD * EE2 + 255) / 256, blk, 0, stream>>>(Wk2, wk2t, HDD, EE2);
  k_transpose_bf16<<<(EEv * EE2 + 255) / 256, blk, 0, stream>>>(Wv, wvt, EEv, EE2);
  k_transpose_bf16<<<(EE2 * EEv + 255) / 256, blk, 0, stream>>>(Wo, wot, EE2, EEv);

  auto grid_for = [](int M, int N) { return dim3((M / 64) * (N / 128)); };

  // low-rank projections (down to head_dim, kept bf16 for next GEMM's A operand)
  k_wmma_gemm<ACT_NONE, true ><<<grid_for(BNv, HDD), blk, 0, stream>>>(xbf, wq1t, tq, BNv, HDD, EEv);
  k_wmma_gemm<ACT_NONE, true ><<<grid_for(BNv, HDD), blk, 0, stream>>>(xbf, wk1t, tk, BNv, HDD, EEv);
  // expansions with fused activations: q = silu(.), a = sigmoid(sigmoid(.))
  k_wmma_gemm<ACT_SILU, false><<<grid_for(BNv, EE2), blk, 0, stream>>>(tq, wq2t, Qb, BNv, EE2, HDD);
  k_wmma_gemm<ACT_GATE, false><<<grid_for(BNv, EE2), blk, 0, stream>>>(tk, wk2t, Ab, BNv, EE2, HDD);
  // v = x @ Wv
  k_wmma_gemm<ACT_NONE, false><<<grid_for(BNv, EE2), blk, 0, stream>>>(xbf, wvt, Vb, BNv, EE2, EEv);

  // elementwise gated recurrence + gating (Qb <- q*h)
  k_hgru_scan<<<(BB * EE2 + 255) / 256, blk, 0, stream>>>(Qb, Ab, Vb);
  // RMSNorm -> bf16
  k_rmsnorm_bf16<<<BNv, blk, 0, stream>>>(Qb, nw, obf);
  // final projection: d_out = out @ Wo  (f32 output)
  k_wmma_gemm<ACT_NONE, false><<<grid_for(BNv, EEv), blk, 0, stream>>>(obf, wot, d_out, BNv, EEv, EE2);
}
